// CriticNetwork_54305566490685
// MI455X (gfx1250) — matrix operand order repaired and checked by reading, not compiled, not run on hardware
//
#include <hip/hip_runtime.h>
#include <math.h>

// ---------------------------------------------------------------------------
// CDNA5 (gfx1250) fused critic network.
// All large GEMMs run through v_wmma_f32_16x16x32_bf16 (wave32 WMMA).
//   - weights converted to bf16 and pre-transposed to [N][K]
//   - 128x64 block tile, 4 waves, 2 row-stripes per wave (8 WMMAs / K-step)
//   - double-buffered LDS staging: one barrier per K-step, global->LDS of the
//     next tile overlaps the current WMMA batch
// ---------------------------------------------------------------------------

typedef __attribute__((ext_vector_type(16))) __bf16 v16bf;
typedef __attribute__((ext_vector_type(8)))  __bf16 v8bf;
typedef __attribute__((ext_vector_type(8)))  float  v8f;

union FragU {
    v16bf v;
    struct { v8bf lo; v8bf hi; } h;
};

#define BATCH 32768
#define KNB   8
#define TM 128          // block tile rows
#define TN 64           // block tile cols
#define TK 32           // k-step
#define LDSS 40         // LDS row stride (elements): 80B -> 16B-aligned chunks

// ---------------------------------------------------------------------------
// weight convert + transpose:  src (Kd x N, f32) -> dst (N x Kd, bf16)
// ---------------------------------------------------------------------------
__global__ void cvt_transpose_kernel(const float* __restrict__ src,
                                     __bf16* __restrict__ dst,
                                     int Kd, int N) {
    int i = blockIdx.x * blockDim.x + threadIdx.x;
    if (i < Kd * N) {
        int k = i / N;
        int n = i - k * N;
        dst[(size_t)n * Kd + k] = (__bf16)src[i];
    }
}

// ---------------------------------------------------------------------------
// Generic WMMA GEMM:  C(M,N) = act(A(M,KD) @ W(KD,N) + bias) -> bf16 out
//   Wt: pre-transposed bf16 weight, row-major N x KD.
//   Block = 128 threads = 4 wave32s. Tile 128x64. Each wave: 32 rows
//   (2 stripes of 16) x 4 N-subtiles -> 8 v8f accumulators.
//   KD must be a multiple of 16 (16-chunk fully valid or fully zero).
// ---------------------------------------------------------------------------
template<int KD, int LDA, int LDC, bool AF32, bool RELU, bool BIAS>
__global__ __launch_bounds__(128)
void gemm_wmma_kernel(const float*  __restrict__ Af,
                      const __bf16* __restrict__ Ah,
                      const __bf16* __restrict__ Wt,
                      const float*  __restrict__ bias,
                      __bf16* __restrict__ C) {
    static_assert(KD % 16 == 0, "KD must be a multiple of 16");

    __shared__ __bf16 As[2][TM][LDSS];   // [buf][m][k]
    __shared__ __bf16 Bs[2][TN][LDSS];   // [buf][n][k]

    const int tid  = threadIdx.x;
    const int lane = tid & 31;
    const int w    = tid >> 5;                  // wave id 0..3
    const int m0   = blockIdx.x * TM;
    const int n0   = blockIdx.y * TN;

    v8f acc[2][4];
    for (int s = 0; s < 2; ++s)
        for (int t = 0; t < 4; ++t)
            for (int i = 0; i < 8; ++i) acc[s][t][i] = 0.0f;

    // shared thread->(row, 16-chunk) map for staging
    const int lrow  = tid >> 1;   // 0..63
    const int lhalf = tid & 1;    // which 16-element k-chunk

    // stage one K-tile (A: 128x32 in two 64-row halves, B: 64x32) into buffer
    auto stage = [&](int buf, int kk) {
        const bool v16 = (kk + lhalf * 16 + 16) <= KD;
        // ---- A (two rows per thread) ----
        for (int r2 = 0; r2 < 2; ++r2) {
            const int row = lrow + r2 * 64;
            v8bf a0, a1;
            if (v16) {
                if (AF32) {
                    const float* ab = Af + (size_t)(m0 + row) * LDA + kk + lhalf * 16;
                    float4 f0 = *(const float4*)(ab + 0);
                    float4 f1 = *(const float4*)(ab + 4);
                    float4 f2 = *(const float4*)(ab + 8);
                    float4 f3 = *(const float4*)(ab + 12);
                    a0[0] = (__bf16)f0.x; a0[1] = (__bf16)f0.y;
                    a0[2] = (__bf16)f0.z; a0[3] = (__bf16)f0.w;
                    a0[4] = (__bf16)f1.x; a0[5] = (__bf16)f1.y;
                    a0[6] = (__bf16)f1.z; a0[7] = (__bf16)f1.w;
                    a1[0] = (__bf16)f2.x; a1[1] = (__bf16)f2.y;
                    a1[2] = (__bf16)f2.z; a1[3] = (__bf16)f2.w;
                    a1[4] = (__bf16)f3.x; a1[5] = (__bf16)f3.y;
                    a1[6] = (__bf16)f3.z; a1[7] = (__bf16)f3.w;
                } else {
                    const __bf16* ab = Ah + (size_t)(m0 + row) * LDA + kk + lhalf * 16;
                    a0 = *(const v8bf*)(ab);
                    a1 = *(const v8bf*)(ab + 8);
                }
            } else {
                for (int e = 0; e < 8; ++e) { a0[e] = (__bf16)0.0f; a1[e] = (__bf16)0.0f; }
            }
            *(v8bf*)&As[buf][row][lhalf * 16]     = a0;
            *(v8bf*)&As[buf][row][lhalf * 16 + 8] = a1;
        }
        // ---- B from pre-transposed weights ----
        v8bf b0, b1;
        if (v16) {
            const __bf16* wb = Wt + (size_t)(n0 + lrow) * KD + kk + lhalf * 16;
            b0 = *(const v8bf*)(wb);
            b1 = *(const v8bf*)(wb + 8);
        } else {
            for (int e = 0; e < 8; ++e) { b0[e] = (__bf16)0.0f; b1[e] = (__bf16)0.0f; }
        }
        *(v8bf*)&Bs[buf][lrow][lhalf * 16]     = b0;
        *(v8bf*)&Bs[buf][lrow][lhalf * 16 + 8] = b1;
    };

    stage(0, 0);
    __syncthreads();

    int buf = 0;
    for (int kk = 0; kk < KD; kk += TK) {
        // prefetch next K-tile into the other buffer (overlaps WMMA batch)
        if (kk + TK < KD) stage(buf ^ 1, kk + TK);

        // ---- load ALL fragments first, then back-to-back WMMAs ----
        // A 16x32: lanes 0-15 K=0..15, lanes 16-31 K=16..31
        FragU af[2];
        const int ak = (lane >> 4) * 16;
        for (int s = 0; s < 2; ++s) {
            const int ar = w * 32 + s * 16 + (lane & 15);
            af[s].h.lo = *(const v8bf*)&As[buf][ar][ak];
            af[s].h.hi = *(const v8bf*)&As[buf][ar][ak + 8];
        }
        // B 32x16: lanes 0-15 K=0..15, lanes 16-31 K=16..31
        FragU bfr[4];
        const int bkb = (lane >> 4) * 16;
        for (int t = 0; t < 4; ++t) {
            const int bn = t * 16 + (lane & 15);
            bfr[t].h.lo = *(const v8bf*)&Bs[buf][bn][bkb];
            bfr[t].h.hi = *(const v8bf*)&Bs[buf][bn][bkb + 8];
        }
        for (int t = 0; t < 4; ++t)
            for (int s = 0; s < 2; ++s)
                acc[s][t] = __builtin_amdgcn_wmma_f32_16x16x32_bf16(
                    false, af[s].v, false, bfr[t].v, (short)0, acc[s][t], false, false);

        __syncthreads();   // single barrier per K-step (double-buffered)
        buf ^= 1;
    }

    // ---- writeback: C layout lane 0-15 -> M=i, lane 16-31 -> M=i+8 ----
    for (int s = 0; s < 2; ++s) {
        const int crow0 = m0 + w * 32 + s * 16 + ((lane >> 4) * 8);
        for (int t = 0; t < 4; ++t) {
            const int col = n0 + t * 16 + (lane & 15);
            const float bv = BIAS ? bias[col] : 0.0f;
            for (int i = 0; i < 8; ++i) {
                float x = acc[s][t][i] + bv;
                if (RELU) x = fmaxf(x, 0.0f);
                C[(size_t)(crow0 + i) * LDC + col] = (__bf16)x;
            }
        }
    }
}

// ---------------------------------------------------------------------------
// mask[b,k] = (sum(state2[b,k,:]) != 0)   (mean != 0 <=> sum != 0)
// one wave32 per row of 384 f32
// ---------------------------------------------------------------------------
__global__ __launch_bounds__(256)
void mask_kernel(const float* __restrict__ s2, float* __restrict__ mask) {
    const int row  = (blockIdx.x * blockDim.x + threadIdx.x) >> 5;
    const int lane = threadIdx.x & 31;
    const float* p = s2 + (size_t)row * 384;
    float s = 0.0f;
    for (int i = lane; i < 384; i += 32) s += p[i];
    for (int off = 16; off; off >>= 1) s += __shfl_xor(s, off, 32);
    if (lane == 0) mask[row] = (s != 0.0f) ? 1.0f : 0.0f;
}

// ---------------------------------------------------------------------------
// masked attention over K=8 neighbors; one wave32 per batch row.
// score_j = (k_j . q)/16 ; softmax over unmasked j ; v_att = sum alpha_j v_j
// writes bf16 into cc[:, 512:768]
// ---------------------------------------------------------------------------
__global__ __launch_bounds__(256)
void attn_kernel(const __bf16* __restrict__ q,
                 const __bf16* __restrict__ k,
                 const __bf16* __restrict__ v,
                 const float*  __restrict__ mask,
                 __bf16* __restrict__ cc) {
    const int b    = (blockIdx.x * blockDim.x + threadIdx.x) >> 5;
    const int lane = threadIdx.x & 31;

    float qv[8];
    {
        v8bf x = *(const v8bf*)(q + (size_t)b * 256 + lane * 8);
        for (int e = 0; e < 8; ++e) qv[e] = (float)x[e];
    }

    float sc[8];
    for (int j = 0; j < 8; ++j) {
        v8bf kv = *(const v8bf*)(k + ((size_t)b * KNB + j) * 256 + lane * 8);
        float p = 0.0f;
        for (int e = 0; e < 8; ++e) p += (float)kv[e] * qv[e];
        for (int off = 16; off; off >>= 1) p += __shfl_xor(p, off, 32);
        sc[j] = (mask[b * KNB + j] != 0.0f) ? (p * 0.0625f) : -3.0e38f;
    }

    float m = sc[0];
    for (int j = 1; j < 8; ++j) m = fmaxf(m, sc[j]);
    float al[8], den = 0.0f;
    for (int j = 0; j < 8; ++j) {
        float e = (sc[j] > -1.0e38f) ? __expf(sc[j] - m) : 0.0f;
        al[j] = e;
        den += e;
    }
    const float rden = (den > 0.0f) ? (1.0f / den) : 0.0f;

    float o[8];
    for (int e = 0; e < 8; ++e) o[e] = 0.0f;
    for (int j = 0; j < 8; ++j) {
        const float a = al[j] * rden;
        v8bf vv = *(const v8bf*)(v + ((size_t)b * KNB + j) * 256 + lane * 8);
        for (int e = 0; e < 8; ++e) o[e] += a * (float)vv[e];
    }
    __bf16* out = cc + (size_t)b * 768 + 512 + lane * 8;
    for (int e = 0; e < 8; ++e) out[e] = (__bf16)o[e];
}

// ---------------------------------------------------------------------------
// final 64-dot:  out[b] = hidden[b,:] . W_j2 + b_j2
// ---------------------------------------------------------------------------
__global__ __launch_bounds__(256)
void qval_kernel(const __bf16* __restrict__ hid,
                 const float* __restrict__ Wj2, const float* __restrict__ bj2,
                 float* __restrict__ out) {
    const int r = blockIdx.x * blockDim.x + threadIdx.x;
    const __bf16* h = hid + (size_t)r * 64;
    float a = bj2[0];
    for (int i = 0; i < 64; ++i) a += (float)h[i] * Wj2[i];
    out[r] = a;
}

// ---------------------------------------------------------------------------
// host launcher
// ---------------------------------------------------------------------------
extern "C" void kernel_launch(void* const* d_in, const int* in_sizes, int n_in,
                              void* d_out, int out_size, void* d_ws, size_t ws_size,
                              hipStream_t stream) {
    (void)in_sizes; (void)n_in; (void)out_size; (void)ws_size;

    const float* state0 = (const float*)d_in[0];
    const float* state1 = (const float*)d_in[1];
    const float* state2 = (const float*)d_in[2];
    const float* W_own  = (const float*)d_in[3];
    const float* b_own  = (const float*)d_in[4];
    const float* W_env  = (const float*)d_in[5];
    const float* b_env  = (const float*)d_in[6];
    const float* W_sur  = (const float*)d_in[7];
    const float* b_sur  = (const float*)d_in[8];
    const float* Wq     = (const float*)d_in[9];
    const float* Wk     = (const float*)d_in[10];
    const float* Wv     = (const float*)d_in[11];
    // d_in[12] (Wcq), d_in[13] (Wck): dead code — softmax over a length-1 axis is 1.
    const float* Wcv    = (const float*)d_in[14];
    const float* W_out  = (const float*)d_in[15];
    const float* b_out  = (const float*)d_in[16];
    const float* W_j1   = (const float*)d_in[17];
    const float* b_j1   = (const float*)d_in[18];
    const float* W_j2   = (const float*)d_in[19];
    const float* b_j2   = (const float*)d_in[20];
    float* out = (float*)d_out;

    const size_t B  = BATCH;
    const size_t BK = (size_t)BATCH * KNB;

    // ---- workspace carve-up ----
    char* base = (char*)d_ws;
    size_t off = 0;
    auto carve = [&](size_t bytes) -> char* {
        off = (off + 255) & ~(size_t)255;
        char* p = base + off;
        off += bytes;
        return p;
    };
    __bf16* wsur  = (__bf16*)carve(BK * 256 * 2);
    __bf16* kbf   = (__bf16*)carve(BK * 256 * 2);
    __bf16* vbf   = (__bf16*)carve(BK * 256 * 2);
    __bf16* qbf   = (__bf16*)carve(B * 256 * 2);
    __bf16* cc    = (__bf16*)carve(B * 768 * 2);
    __bf16* catv  = (__bf16*)carve(B * 768 * 2);
    __bf16* multi = (__bf16*)carve(B * 128 * 2);
    __bf16* hid   = (__bf16*)carve(B * 64 * 2);
    float*  maskb = (float*)carve(BK * 4);
    // pre-transposed bf16 weights: Wt[N][Kd]
    __bf16* wtOwn = (__bf16*)carve(256 * 80  * 2);
    __bf16* wtEnv = (__bf16*)carve(256 * 160 * 2);
    __bf16* wtSur = (__bf16*)carve(256 * 384 * 2);
    __bf16* wtQ   = (__bf16*)carve(256 * 256 * 2);
    __bf16* wtK   = (__bf16*)carve(256 * 256 * 2);
    __bf16* wtV   = (__bf16*)carve(256 * 256 * 2);
    __bf16* wtCv  = (__bf16*)carve(256 * 256 * 2);
    __bf16* wtOut = (__bf16*)carve(128 * 768 * 2);
    __bf16* wtJ1  = (__bf16*)carve(64  * 128 * 2);

    // ---- weight convert+transpose (tiny, L2-resident afterwards) ----
    auto cvt = [&](const float* s, __bf16* d, int Kd, int N) {
        int n = Kd * N;
        cvt_transpose_kernel<<<(n + 255) / 256, 256, 0, stream>>>(s, d, Kd, N);
    };
    cvt(W_own, wtOwn, 80, 256);
    cvt(W_env, wtEnv, 160, 256);
    cvt(W_sur, wtSur, 384, 256);
    cvt(Wq,    wtQ,   256, 256);
    cvt(Wk,    wtK,   256, 256);
    cvt(Wv,    wtV,   256, 256);
    cvt(Wcv,   wtCv,  256, 256);
    cvt(W_out, wtOut, 768, 128);
    cvt(W_j1,  wtJ1,  128, 64);

    // ---- neighbor validity mask from state2 ----
    mask_kernel<<<(unsigned)(BK / 8), 256, 0, stream>>>(state2, maskb);

    // ---- input projections (f32 A -> bf16 out, fused bias+ReLU) ----
    gemm_wmma_kernel<384, 384, 256, true, true, true>
        <<<dim3((unsigned)(BK / TM), 256 / TN), 128, 0, stream>>>(
            state2, nullptr, wtSur, b_sur, wsur);
    gemm_wmma_kernel<80, 80, 768, true, true, true>
        <<<dim3((unsigned)(B / TM), 256 / TN), 128, 0, stream>>>(
            state0, nullptr, wtOwn, b_own, cc /*cols 0:256*/);
    gemm_wmma_kernel<160, 160, 768, true, true, true>
        <<<dim3((unsigned)(B / TM), 256 / TN), 128, 0, stream>>>(
            state1, nullptr, wtEnv, b_env, cc + 256 /*cols 256:512*/);

    // ---- Q/K/V projections (bf16 A) ----
    gemm_wmma_kernel<256, 768, 256, false, false, false>
        <<<dim3((unsigned)(B / TM), 256 / TN), 128, 0, stream>>>(
            nullptr, cc, wtQ, nullptr, qbf);
    gemm_wmma_kernel<256, 256, 256, false, false, false>
        <<<dim3((unsigned)(BK / TM), 256 / TN), 128, 0, stream>>>(
            nullptr, wsur, wtK, nullptr, kbf);
    gemm_wmma_kernel<256, 256, 256, false, false, false>
        <<<dim3((unsigned)(BK / TM), 256 / TN), 128, 0, stream>>>(
            nullptr, wsur, wtV, nullptr, vbf);

    // ---- masked softmax attention over K=8; fills cc[:,512:768] ----
    attn_kernel<<<(unsigned)(B / 8), 256, 0, stream>>>(qbf, kbf, vbf, maskb, cc);

    // ---- "multi-head attention" with seq_len=1: softmax==1 -> just comV.
    //      cc.reshape(3B,256) @ Wcv  (one flat GEMM covering all 3 heads) ----
    gemm_wmma_kernel<256, 256, 256, false, false, false>
        <<<dim3((unsigned)(3 * B / TM), 256 / TN), 128, 0, stream>>>(
            nullptr, cc, wtCv, nullptr, catv);

    // ---- multi_att_out: (B,768) @ (768,128) + b_out ----
    gemm_wmma_kernel<768, 768, 128, false, false, true>
        <<<dim3((unsigned)(B / TM), 128 / TN), 128, 0, stream>>>(
            nullptr, catv, wtOut, b_out, multi);

    // ---- judgement_fc layer 1: relu((B,128)@(128,64)+b) ----
    gemm_wmma_kernel<128, 128, 64, false, true, true>
        <<<dim3((unsigned)(B / TM), 64 / TN), 128, 0, stream>>>(
            nullptr, multi, wtJ1, b_j1, hid);

    // ---- final dot -> f32 output (B,1,1) ----
    qval_kernel<<<(unsigned)(B / 256), 256, 0, stream>>>(hid, W_j2, b_j2, out);
}
